// HypLinear_10969346474537
// MI455X (gfx1250) — compile-verified, matrix operand order checked by reading, not run on hardware
//
#include <hip/hip_runtime.h>

// ---------------------------------------------------------------------------
// HypLinear for MI455X (gfx1250, wave32, WMMA)
//   hyp_w = proj(expmap0(weight, c), c)            -> prep_w kernel (bf16 hi/lo split)
//   mx    = x @ hyp_w^T                            -> bf16 WMMA, fp32 accum, 3-term split
//   res   = proj(mobius_matvec epilogue)           -> per-row scalars, fused
//   out   = proj(mobius_add(res, hyp_b, c), c)     -> analytic norms, fused
// ---------------------------------------------------------------------------

typedef __attribute__((ext_vector_type(16))) __bf16 v16bf;
typedef __attribute__((ext_vector_type(8)))  __bf16 v8bf;
typedef __attribute__((ext_vector_type(8)))  float  v8f;

#define KDIM 1024
#define ODIM 1024
#define BALL_EPS 4e-3f
#define MIN_NORM 1e-15f

__device__ __forceinline__ float artanh_clip(float z) {
    z = fminf(fmaxf(z, -1.0f + 1e-7f), 1.0f - 1e-7f);
    return 0.5f * logf((1.0f + z) / (1.0f - z));
}

// ---------------------------------------------------------------------------
// prep_w: one block per weight row. Row-norm -> expmap0 + proj scale ->
// write scaled row as bf16 hi + bf16 residual lo.
// ---------------------------------------------------------------------------
__global__ __launch_bounds__(256) void prep_w_kernel(const float* __restrict__ w,
                                                     const float* __restrict__ cptr,
                                                     __bf16* __restrict__ wh,
                                                     __bf16* __restrict__ wl) {
    const int row = blockIdx.x;
    const float* wr = w + (size_t)row * KDIM;
    float ss = 0.0f;
    for (int i = threadIdx.x; i < KDIM; i += 256) {
        float v = wr[i];
        ss += v * v;
    }
    __shared__ float sred[8];
    __shared__ float s_scale;
    #pragma unroll
    for (int m = 16; m >= 1; m >>= 1) ss += __shfl_xor(ss, m, 32);
    if ((threadIdx.x & 31) == 0) sred[threadIdx.x >> 5] = ss;
    __syncthreads();
    if (threadIdx.x == 0) {
        float tot = 0.0f;
        #pragma unroll
        for (int wv = 0; wv < 8; ++wv) tot += sred[wv];
        float cc = cptr[0];
        float sc = sqrtf(cc);
        float n  = fmaxf(sqrtf(tot), MIN_NORM);
        float th = tanhf(sc * n);
        float scale = th / (sc * n);            // expmap0
        float rn    = th / sc;                  // resulting norm
        float maxn  = (1.0f - BALL_EPS) / sc;   // proj clamp
        if (rn > maxn) scale *= maxn / rn;
        s_scale = scale;
    }
    __syncthreads();
    float sc = s_scale;
    for (int i = threadIdx.x; i < KDIM; i += 256) {
        float v = wr[i] * sc;
        __bf16 h = (__bf16)v;
        wh[(size_t)row * KDIM + i] = h;
        wl[(size_t)row * KDIM + i] = (__bf16)(v - (float)h);
    }
}

// ---------------------------------------------------------------------------
// prep_b: hyp_b = proj(expmap0(bias, c), c), plus ||hyp_b||^2 scalar.
// ---------------------------------------------------------------------------
__global__ __launch_bounds__(256) void prep_b_kernel(const float* __restrict__ bias,
                                                     const float* __restrict__ cptr,
                                                     float* __restrict__ hb,
                                                     float* __restrict__ b2out) {
    float ss = 0.0f;
    for (int i = threadIdx.x; i < ODIM; i += 256) {
        float v = bias[i];
        ss += v * v;
    }
    __shared__ float sred[8];
    __shared__ float s_scale;
    #pragma unroll
    for (int m = 16; m >= 1; m >>= 1) ss += __shfl_xor(ss, m, 32);
    if ((threadIdx.x & 31) == 0) sred[threadIdx.x >> 5] = ss;
    __syncthreads();
    if (threadIdx.x == 0) {
        float tot = 0.0f;
        #pragma unroll
        for (int wv = 0; wv < 8; ++wv) tot += sred[wv];
        float cc = cptr[0];
        float sc = sqrtf(cc);
        float n  = fmaxf(sqrtf(tot), MIN_NORM);
        float th = tanhf(sc * n);
        float scale = th / (sc * n);
        float rn    = th / sc;
        float maxn  = (1.0f - BALL_EPS) / sc;
        if (rn > maxn) { scale *= maxn / rn; rn = maxn; }
        s_scale = scale;
        b2out[0] = rn * rn;                     // ||hyp_b||^2
    }
    __syncthreads();
    float sc = s_scale;
    for (int i = threadIdx.x; i < ODIM; i += 256) hb[i] = bias[i] * sc;
}

// ---------------------------------------------------------------------------
// Fused GEMM + hyperbolic epilogue.
// Block = 16 output rows x all 1024 columns. 8 waves, each wave owns 8
// 16x16 column tiles. bf16 WMMA with hi/lo split: D += Ah*Bh + Ah*Bl + Al*Bh.
// ---------------------------------------------------------------------------
__global__ __launch_bounds__(256) void hyp_gemm_kernel(const float* __restrict__ x,
                                                       const __bf16* __restrict__ wh,
                                                       const __bf16* __restrict__ wl,
                                                       const float* __restrict__ hb,
                                                       const float* __restrict__ b2ptr,
                                                       const float* __restrict__ cptr,
                                                       float* __restrict__ out) {
    const int lane    = threadIdx.x & 31;
    const int wave    = threadIdx.x >> 5;
    const int lr      = lane & 15;       // row-in-tile (A) / col-in-tile (B,C)
    const int halfsel = lane >> 4;       // lane half
    const int kbase   = halfsel * 8;     // K sub-offset per WMMA 16-bit layout
    const int row16   = blockIdx.x * 16;
    const int colbase = wave * 128;

    const float* xrow = x + (size_t)(row16 + lr) * KDIM;

    v8f acc[8];
    #pragma unroll
    for (int t = 0; t < 8; ++t)
        #pragma unroll
        for (int r = 0; r < 8; ++r) acc[t][r] = 0.0f;

    float xsq = 0.0f;

    for (int k0 = 0; k0 < KDIM; k0 += 32) {
        // ---- A tile: 16 fp32 -> bf16 hi/lo (layout per CDNA5 ISA 16-bit A) ----
        float4 p0 = *(const float4*)(xrow + k0 + kbase + 0);
        float4 p1 = *(const float4*)(xrow + k0 + kbase + 4);
        float4 p2 = *(const float4*)(xrow + k0 + kbase + 16);
        float4 p3 = *(const float4*)(xrow + k0 + kbase + 20);
        float av[16] = {p0.x, p0.y, p0.z, p0.w, p1.x, p1.y, p1.z, p1.w,
                        p2.x, p2.y, p2.z, p2.w, p3.x, p3.y, p3.z, p3.w};
        v16bf ah, al;
        #pragma unroll
        for (int i = 0; i < 16; ++i) {
            float v = av[i];
            __bf16 h = (__bf16)v;
            ah[i] = h;
            al[i] = (__bf16)(v - (float)h);
            xsq += v * v;
        }
        // ---- B tiles: precomputed bf16 hi/lo of hyp_w ([out][k] row-major) ----
        #pragma unroll
        for (int t = 0; t < 8; ++t) {
            size_t boff = (size_t)(colbase + t * 16 + lr) * KDIM + k0 + kbase;
            v8bf h0 = *(const v8bf*)(wh + boff);
            v8bf h1 = *(const v8bf*)(wh + boff + 16);
            v8bf l0 = *(const v8bf*)(wl + boff);
            v8bf l1 = *(const v8bf*)(wl + boff + 16);
            v16bf bh, bl;
            #pragma unroll
            for (int i = 0; i < 8; ++i) {
                bh[i] = h0[i]; bh[8 + i] = h1[i];
                bl[i] = l0[i]; bl[8 + i] = l1[i];
            }
            acc[t] = __builtin_amdgcn_wmma_f32_16x16x32_bf16(false, ah, false, bh,
                                                             (short)0, acc[t], false, false);
            acc[t] = __builtin_amdgcn_wmma_f32_16x16x32_bf16(false, ah, false, bl,
                                                             (short)0, acc[t], false, false);
            acc[t] = __builtin_amdgcn_wmma_f32_16x16x32_bf16(false, al, false, bh,
                                                             (short)0, acc[t], false, false);
        }
    }

    __shared__ float s_xsq[16];
    __shared__ float s_rsq[8][16];
    __shared__ float s_dot[8][16];
    __shared__ float s_P[16];
    __shared__ float s_Q[16];

    // ||x_row||^2: lanes l and l^16 cover complementary K; wave 0 publishes.
    {
        float full = xsq + __shfl_xor(xsq, 16, 32);
        if (wave == 0 && lane < 16) s_xsq[lane] = full;
    }

    // bias values for this wave's columns
    float bv[8];
    #pragma unroll
    for (int t = 0; t < 8; ++t) bv[t] = hb[colbase + t * 16 + lr];

    // per-wave row reductions of ||mx||^2 and mx . hyp_b
    #pragma unroll
    for (int r = 0; r < 8; ++r) {
        float sq = 0.0f, dt = 0.0f;
        #pragma unroll
        for (int t = 0; t < 8; ++t) {
            float v = acc[t][r];
            sq += v * v;
            dt += v * bv[t];
        }
        #pragma unroll
        for (int m = 8; m >= 1; m >>= 1) {
            sq += __shfl_xor(sq, m, 16);
            dt += __shfl_xor(dt, m, 16);
        }
        if (lr == 0) {
            int row = r + halfsel * 8;
            s_rsq[wave][row] = sq;
            s_dot[wave][row] = dt;
        }
    }
    __syncthreads();

    // per-row hyperbolic scalars: out = P[row]*mx + Q[row]*hyp_b
    if (threadIdx.x < 16) {
        int row = threadIdx.x;
        float msq = 0.0f, mdot = 0.0f;
        #pragma unroll
        for (int wv = 0; wv < 8; ++wv) { msq += s_rsq[wv][row]; mdot += s_dot[wv][row]; }
        float cc = cptr[0];
        float sc = sqrtf(cc);
        float xn = fmaxf(sqrtf(s_xsq[row]), MIN_NORM);
        float mn = fmaxf(sqrtf(msq), MIN_NORM);
        float at = artanh_clip(sc * xn);
        float s  = tanhf(mn / xn * at);
        float sclamp = fminf(s, 1.0f - BALL_EPS);      // proj(res) (s >= 0 here)
        float f  = sclamp / (sc * mn);                 // res = f * mx
        float r2 = sclamp * sclamp / cc;               // ||res||^2
        if (msq == 0.0f) { f = 0.0f; r2 = 0.0f; }      // zero-row guard
        float y2 = b2ptr[0];
        float xy = f * mdot;
        float Am = 1.0f + 2.0f * cc * xy + cc * y2;    // coeff on res
        float Bm = 1.0f - cc * r2;                     // coeff on hyp_b
        float den = 1.0f + 2.0f * cc * xy + cc * cc * r2 * y2;
        den = fmaxf(den, MIN_NORM);
        // analytic norm of mobius_add result for final proj
        float n2 = (Am * Am * r2 + 2.0f * Am * Bm * xy + Bm * Bm * y2) / (den * den);
        float nn = fmaxf(sqrtf(n2), MIN_NORM);
        float maxn = (1.0f - BALL_EPS) / sc;
        float ps = (nn > maxn) ? (maxn / nn) : 1.0f;
        s_P[row] = ps * Am * f / den;
        s_Q[row] = ps * Bm / den;
    }
    __syncthreads();

    // store
    #pragma unroll
    for (int t = 0; t < 8; ++t) {
        int col = colbase + t * 16 + lr;
        #pragma unroll
        for (int r = 0; r < 8; ++r) {
            int row = r + halfsel * 8;
            out[(size_t)(row16 + row) * ODIM + col] = s_P[row] * acc[t][r] + s_Q[row] * bv[t];
        }
    }
}

// ---------------------------------------------------------------------------
extern "C" void kernel_launch(void* const* d_in, const int* in_sizes, int n_in,
                              void* d_out, int out_size, void* d_ws, size_t ws_size,
                              hipStream_t stream) {
    const float* x    = (const float*)d_in[0];   // [N, 1024]
    const float* c    = (const float*)d_in[1];   // [1]
    const float* w    = (const float*)d_in[2];   // [1024, 1024]
    const float* bias = (const float*)d_in[3];   // [1024]
    float* out = (float*)d_out;

    const int N = in_sizes[0] / KDIM;

    // workspace layout
    char* ws = (char*)d_ws;
    __bf16* wh = (__bf16*)ws;                                   // 2 MB
    __bf16* wl = (__bf16*)(ws + (size_t)ODIM * KDIM * 2);       // 2 MB
    float*  hb = (float*)(ws + (size_t)ODIM * KDIM * 4);        // 4 KB
    float*  b2 = (float*)(ws + (size_t)ODIM * KDIM * 4 + ODIM * sizeof(float));

    prep_w_kernel<<<ODIM, 256, 0, stream>>>(w, c, wh, wl);
    prep_b_kernel<<<1, 256, 0, stream>>>(bias, c, hb, b2);
    hyp_gemm_kernel<<<N / 16, 256, 0, stream>>>(x, wh, wl, hb, b2, c, out);
}
